// MessagePassingLayer_56229711839297
// MI455X (gfx1250) — matrix operand order repaired and checked by reading, not compiled, not run on hardware
//
#include <hip/hip_runtime.h>
#include <hip/hip_fp16.h>
#include <hip/hip_bf16.h>

typedef __attribute__((ext_vector_type(16))) _Float16 v16h;
typedef __attribute__((ext_vector_type(8)))  _Float16 v8h;
typedef __attribute__((ext_vector_type(8)))  float    v8f;

#define NODES 50000
#define EDGES 800000

// ---------------------------------------------------------------------------
// fp32 -> fp16 conversion
__global__ void cvt_f32_f16(const float* __restrict__ in, _Float16* __restrict__ out, int n) {
    int i = blockIdx.x * blockDim.x + threadIdx.x;
    if (i < n) out[i] = (_Float16)in[i];
}

// ---------------------------------------------------------------------------
// Pack a row-major (K x 128) fp32 weight matrix into WMMA B-fragments (f16).
// Fragment (kt, nt) holds the 32x16 tile  K = kt*32..+31, N = nt*16..+15.
// Lane l, element j:  col = nt*16 + (l&15),  k = kt*32 + (j>>3)*16 + (l>>4)*8 + (j&7).
// Storage: frag-major, ((nt*KT + kt)*32 + lane)*16 + j   (contiguous 32B/lane).
__global__ void pack_weights(const float* __restrict__ W, _Float16* __restrict__ out,
                             int K, int Ntot) {
    int idx = blockIdx.x * blockDim.x + threadIdx.x;
    int total = K * Ntot;
    if (idx >= total) return;
    int j    = idx & 15;
    int lane = (idx >> 4) & 31;
    int frag = idx >> 9;
    int KT   = K >> 5;
    int kt   = frag % KT;
    int nt   = frag / KT;
    int col  = nt * 16 + (lane & 15);
    int k    = kt * 32 + ((j >> 3) * 16) + ((lane >> 4) * 8) + (j & 7);
    out[idx] = (_Float16)W[k * Ntot + col];
}

// ---------------------------------------------------------------------------
// A-fragment chunk loader for the edge concat [h[src] | h[dst] | e]  (width 288)
__device__ __forceinline__ v8h load_chunk_edge(const _Float16* __restrict__ h16,
                                               const _Float16* __restrict__ ea16,
                                               int src, int dst, int e, int c) {
    const _Float16* p;
    if (c < 128)      p = h16 + (size_t)src * 128 + c;
    else if (c < 256) p = h16 + (size_t)dst * 128 + (c - 128);
    else              p = ea16 + (size_t)e * 32 + (c - 256);
    return *(const v8h*)p;
}

// A-fragment chunk loader for the node concat [h | agg] (width 256, both f16)
__device__ __forceinline__ v8h load_chunk_node(const _Float16* __restrict__ h16,
                                               const _Float16* __restrict__ aggH,
                                               int node, int c) {
    const _Float16* p;
    if (c < 128) p = h16 + (size_t)node * 128 + c;
    else         p = aggH + (size_t)node * 128 + (c - 128);
    return *(const v8h*)p;
}

// cooperative global->LDS copy of packed weights (v8h granules)
__device__ __forceinline__ void stage_weights(const _Float16* __restrict__ g,
                                              _Float16* s, int nhalf, int t, int nthreads) {
    const v8h* src = (const v8h*)g;
    v8h*       dstp = (v8h*)s;
    for (int i = t; i < (nhalf >> 3); i += nthreads) dstp[i] = src[i];
}

// ---------------------------------------------------------------------------
// Edge kernel: one wave per 32-edge super-tile (2 M-tiles share each B frag).
// Weights live in LDS (staged once per block); B reads are ds_load_b128 pairs.
//   x   = relu([h_src|h_dst|e] @ Wm1 + bm1)     (2x 16x128, 9 K-steps)
//   msg = relu(x @ Wm2 + bm2)                   (2x 16x128, 4 K-steps)
//   aggH[dst] += msg                            (packed f16 atomics, even lanes)
__global__ __launch_bounds__(128) void edge_kernel(
    const _Float16* __restrict__ h16, const _Float16* __restrict__ ea16,
    const long long* __restrict__ ei,
    const _Float16* __restrict__ W1, const float* __restrict__ b1,
    const _Float16* __restrict__ W2, const float* __restrict__ b2,
    _Float16* __restrict__ aggH, int E_) {
    __shared__ __align__(32) _Float16 sW1[288 * 128];   // 73728 B
    __shared__ __align__(32) _Float16 sW2[128 * 128];   // 32768 B
    __shared__ __align__(32) _Float16 xt[4][2][16 * 128];

    stage_weights(W1, sW1, 288 * 128, threadIdx.x, 128);
    stage_weights(W2, sW2, 128 * 128, threadIdx.x, 128);
    __syncthreads();

    const int lane = threadIdx.x & 31;
    const int wave = threadIdx.x >> 5;
    const int wt   = blockIdx.x * 4 + wave;
    if (wt * 32 >= E_) return;

    const int hi = lane >> 4;
    const int lo = lane & 15;
    const int eb = wt * 32;

    int e[2], src[2], dst[2];
#pragma unroll
    for (int m = 0; m < 2; ++m) {
        e[m]   = eb + m * 16 + lo;
        src[m] = (int)ei[e[m]];
        dst[m] = (int)ei[(long long)E_ + e[m]];
    }

    // ---- layer 1: (2x 16x288) @ (288x128) ----
    v8f acc[2][8];
#pragma unroll
    for (int m = 0; m < 2; ++m)
#pragma unroll
        for (int nt = 0; nt < 8; ++nt) acc[m][nt] = {};

#pragma unroll
    for (int kt = 0; kt < 9; ++kt) {
        union { v16h v; v8h h[2]; } a[2];
#pragma unroll
        for (int m = 0; m < 2; ++m) {
            a[m].h[0] = load_chunk_edge(h16, ea16, src[m], dst[m], e[m], kt * 32 + hi * 8);
            a[m].h[1] = load_chunk_edge(h16, ea16, src[m], dst[m], e[m], kt * 32 + 16 + hi * 8);
        }
#pragma unroll
        for (int nt = 0; nt < 8; ++nt) {
            v16h b = *(const v16h*)(sW1 + ((nt * 9 + kt) * 32 + lane) * 16);
#pragma unroll
            for (int m = 0; m < 2; ++m)
                acc[m][nt] = __builtin_amdgcn_wmma_f32_16x16x32_f16(
                    false, a[m].v, false, b, (short)0, acc[m][nt], false, false);
        }
    }

    // bias + relu, stash x tiles (row-major 16x128 f16) in LDS
#pragma unroll
    for (int m = 0; m < 2; ++m) {
        _Float16* xtile = xt[wave][m];
#pragma unroll
        for (int nt = 0; nt < 8; ++nt) {
            float bv = b1[nt * 16 + lo];
#pragma unroll
            for (int i = 0; i < 8; ++i) {
                float v = acc[m][nt][i] + bv;
                v = v > 0.f ? v : 0.f;
                xtile[(i + 8 * hi) * 128 + nt * 16 + lo] = (_Float16)v;
            }
        }
    }
    asm volatile("s_wait_dscnt 0" ::: "memory");  // cross-lane LDS RAW within wave

    // ---- layer 2: (2x 16x128) @ (128x128) ----
    v8f acc2[2][8];
#pragma unroll
    for (int m = 0; m < 2; ++m)
#pragma unroll
        for (int nt = 0; nt < 8; ++nt) acc2[m][nt] = {};

#pragma unroll
    for (int kt = 0; kt < 4; ++kt) {
        union { v16h v; v8h h[2]; } a[2];
#pragma unroll
        for (int m = 0; m < 2; ++m) {
            const _Float16* xtile = xt[wave][m];
            a[m].h[0] = *(const v8h*)(xtile + lo * 128 + kt * 32 + hi * 8);
            a[m].h[1] = *(const v8h*)(xtile + lo * 128 + kt * 32 + 16 + hi * 8);
        }
#pragma unroll
        for (int nt = 0; nt < 8; ++nt) {
            v16h b = *(const v16h*)(sW2 + ((nt * 4 + kt) * 32 + lane) * 16);
#pragma unroll
            for (int m = 0; m < 2; ++m)
                acc2[m][nt] = __builtin_amdgcn_wmma_f32_16x16x32_f16(
                    false, a[m].v, false, b, (short)0, acc2[m][nt], false, false);
        }
    }

    // bias + relu + packed-f16 scatter-add over destination nodes.
    // Lanes l and l^1 hold adjacent columns of the same edge row; even lanes
    // issue one global_atomic_pk_add_f16 per pair. dst for row r=8*hi+i is
    // fetched cross-lane from lane r (which loaded it) via __shfl.
#pragma unroll
    for (int m = 0; m < 2; ++m) {
#pragma unroll
        for (int nt = 0; nt < 8; ++nt) {
            float bv = b2[nt * 16 + lo];
#pragma unroll
            for (int i = 0; i < 8; ++i) {
                float v = acc2[m][nt][i] + bv;
                v = v > 0.f ? v : 0.f;
                float vn = __shfl_xor(v, 1);            // neighbor column's value
                int   dr = __shfl(dst[m], 8 * hi + i);  // dst node of this row
                if ((lo & 1) == 0) {
                    __half2 p = __floats2half2_rn(v, vn);
                    atomicAdd((__half2*)(aggH + (size_t)dr * 128 + nt * 16 + lo), p);
                }
            }
        }
    }
}

// ---------------------------------------------------------------------------
// Node kernel: one wave per 16-node tile, weights staged in LDS.
//   u     = relu([h | agg] @ Wu1 + bu1)
//   h_new = h + u @ Wu2 + bu2
__global__ __launch_bounds__(128) void node_kernel(
    const _Float16* __restrict__ h16, const _Float16* __restrict__ aggH,
    const _Float16* __restrict__ W1, const float* __restrict__ b1,
    const _Float16* __restrict__ W2, const float* __restrict__ b2,
    const float* __restrict__ h, float* __restrict__ out, int N_) {
    __shared__ __align__(32) _Float16 sU1[256 * 128];   // 65536 B
    __shared__ __align__(32) _Float16 sU2[128 * 128];   // 32768 B
    __shared__ __align__(32) _Float16 ut[4][16 * 128];

    stage_weights(W1, sU1, 256 * 128, threadIdx.x, 128);
    stage_weights(W2, sU2, 128 * 128, threadIdx.x, 128);
    __syncthreads();

    const int lane = threadIdx.x & 31;
    const int wave = threadIdx.x >> 5;
    const int tile = blockIdx.x * 4 + wave;
    if (tile * 16 >= N_) return;

    const int hi = lane >> 4;
    const int lo = lane & 15;
    const int nb = tile * 16;
    const int node = nb + lo;

    // ---- layer 1: (16x256) @ (256x128) ----
    v8f acc[8];
#pragma unroll
    for (int nt = 0; nt < 8; ++nt) acc[nt] = {};

#pragma unroll
    for (int kt = 0; kt < 8; ++kt) {
        union { v16h v; v8h h[2]; } a;
        a.h[0] = load_chunk_node(h16, aggH, node, kt * 32 + hi * 8);
        a.h[1] = load_chunk_node(h16, aggH, node, kt * 32 + 16 + hi * 8);
#pragma unroll
        for (int nt = 0; nt < 8; ++nt) {
            v16h b = *(const v16h*)(sU1 + ((nt * 8 + kt) * 32 + lane) * 16);
            acc[nt] = __builtin_amdgcn_wmma_f32_16x16x32_f16(
                false, a.v, false, b, (short)0, acc[nt], false, false);
        }
    }

    _Float16* utile = ut[wave];
#pragma unroll
    for (int nt = 0; nt < 8; ++nt) {
        float bv = b1[nt * 16 + lo];
#pragma unroll
        for (int i = 0; i < 8; ++i) {
            float v = acc[nt][i] + bv;
            v = v > 0.f ? v : 0.f;
            utile[(i + 8 * hi) * 128 + nt * 16 + lo] = (_Float16)v;
        }
    }
    asm volatile("s_wait_dscnt 0" ::: "memory");

    // ---- layer 2: (16x128) @ (128x128), residual + bias ----
    v8f acc2[8];
#pragma unroll
    for (int nt = 0; nt < 8; ++nt) acc2[nt] = {};

#pragma unroll
    for (int kt = 0; kt < 4; ++kt) {
        union { v16h v; v8h h[2]; } a;
        a.h[0] = *(const v8h*)(utile + lo * 128 + kt * 32 + hi * 8);
        a.h[1] = *(const v8h*)(utile + lo * 128 + kt * 32 + 16 + hi * 8);
#pragma unroll
        for (int nt = 0; nt < 8; ++nt) {
            v16h b = *(const v16h*)(sU2 + ((nt * 4 + kt) * 32 + lane) * 16);
            acc2[nt] = __builtin_amdgcn_wmma_f32_16x16x32_f16(
                false, a.v, false, b, (short)0, acc2[nt], false, false);
        }
    }

#pragma unroll
    for (int nt = 0; nt < 8; ++nt) {
        int col = nt * 16 + lo;
        float bv = b2[col];
#pragma unroll
        for (int i = 0; i < 8; ++i) {
            int nrow = nb + 8 * hi + i;
            size_t o = (size_t)nrow * 128 + col;
            out[o] = h[o] + acc2[nt][i] + bv;
        }
    }
}

// ---------------------------------------------------------------------------
extern "C" void kernel_launch(void* const* d_in, const int* in_sizes, int n_in,
                              void* d_out, int out_size, void* d_ws, size_t ws_size,
                              hipStream_t stream) {
    const float*     h    = (const float*)d_in[0];
    const float*     ea   = (const float*)d_in[1];
    const float*     Wm1  = (const float*)d_in[2];
    const float*     bm1  = (const float*)d_in[3];
    const float*     Wm2  = (const float*)d_in[4];
    const float*     bm2  = (const float*)d_in[5];
    const float*     Wu1  = (const float*)d_in[6];
    const float*     bu1  = (const float*)d_in[7];
    const float*     Wu2  = (const float*)d_in[8];
    const float*     bu2  = (const float*)d_in[9];
    const long long* ei   = (const long long*)d_in[10];
    float*           out  = (float*)d_out;

    const int N_ = NODES, E_ = EDGES;

    // workspace carve-up
    char* ws = (char*)d_ws;
    _Float16* aggH = (_Float16*)ws;            size_t off = (size_t)N_ * 128 * 2;
    _Float16* h16  = (_Float16*)(ws + off);    off += (size_t)N_ * 128 * 2;
    _Float16* ea16 = (_Float16*)(ws + off);    off += (size_t)E_ * 32 * 2;
    _Float16* W1p  = (_Float16*)(ws + off);    off += (size_t)288 * 128 * 2;
    _Float16* W2p  = (_Float16*)(ws + off);    off += (size_t)128 * 128 * 2;
    _Float16* U1p  = (_Float16*)(ws + off);    off += (size_t)256 * 128 * 2;
    _Float16* U2p  = (_Float16*)(ws + off);    off += (size_t)128 * 128 * 2;

    hipMemsetAsync(aggH, 0, (size_t)N_ * 128 * 2, stream);

    int nh = N_ * 128, ne = E_ * 32;
    cvt_f32_f16<<<(nh + 255) / 256, 256, 0, stream>>>(h, h16, nh);
    cvt_f32_f16<<<(ne + 255) / 256, 256, 0, stream>>>(ea, ea16, ne);
    pack_weights<<<(288 * 128 + 255) / 256, 256, 0, stream>>>(Wm1, W1p, 288, 128);
    pack_weights<<<(128 * 128 + 255) / 256, 256, 0, stream>>>(Wm2, W2p, 128, 128);
    pack_weights<<<(256 * 128 + 255) / 256, 256, 0, stream>>>(Wu1, U1p, 256, 128);
    pack_weights<<<(128 * 128 + 255) / 256, 256, 0, stream>>>(Wu2, U2p, 128, 128);

    int super = E_ / 32;                        // 25000 32-edge super-tiles
    int ntiles = (N_ + 15) / 16;                // 3125
    edge_kernel<<<(super + 3) / 4, 128, 0, stream>>>(h16, ea16, ei, W1p, bm1, W2p, bm2, aggH, E_);
    node_kernel<<<(ntiles + 3) / 4, 128, 0, stream>>>(h16, aggH, U1p, bu1, U2p, bu2, h, out, N_);
}
